// Attention_5841155522604
// MI455X (gfx1250) — matrix operand order repaired and checked by reading, not compiled, not run on hardware
//
#include <hip/hip_runtime.h>
#include <stdint.h>

// ---------------------------------------------------------------------------
// Types / fragment helpers for gfx1250 WMMA (wave32, 16x16x32 bf16 -> f32)
// ---------------------------------------------------------------------------
typedef __attribute__((ext_vector_type(16))) __bf16        v16bf;
typedef __attribute__((ext_vector_type(8)))  float         v8f;
typedef __attribute__((ext_vector_type(4)))  unsigned int  u32x4;

union FragB16 { v16bf v; u32x4 q[2]; };

__device__ __forceinline__ unsigned short f2bf(float x) {
  unsigned int u = __float_as_uint(x);
  u += 0x7FFFu + ((u >> 16) & 1u);            // round-to-nearest-even
  return (unsigned short)(u >> 16);
}

// A-matrix 16x32 bf16 fragment from row-major src (stride ld halves).
// Lane L<16: row L, K=0..7 & 16..23 ; lane L+16: row L, K=8..15 & 24..31.
__device__ __forceinline__ v16bf load_frag_a16(const unsigned short* p, int ld) {
  const int lane = threadIdx.x & 31;
  const unsigned short* r = p + (size_t)(lane & 15) * ld + ((lane >> 4) << 3);
  FragB16 f;
  f.q[0] = *reinterpret_cast<const u32x4*>(r);
  f.q[1] = *reinterpret_cast<const u32x4*>(r + 16);
  return f.v;
}

// B-matrix 32x16 bf16 fragment from row-major (K-major) src (stride ld halves).
// Lane L<16: col L, K=0..7 & 16..23 ; lane L+16: col L, K=8..15 & 24..31.
__device__ __forceinline__ v16bf load_frag_b16(const unsigned short* p, int ld) {
  const int lane = threadIdx.x & 31;
  const unsigned short* r = p + (lane & 15) + (size_t)((lane >> 4) << 3) * ld;
  union { v16bf v; unsigned short u[16]; } f;
#pragma unroll
  for (int i = 0; i < 8; ++i) {
    f.u[i] = r[(size_t)i * ld];
    f.u[8 + i] = r[(size_t)(16 + i) * ld];
  }
  return f.v;
}

__device__ __forceinline__ v8f wmma_bf16(v16bf a, v16bf b, v8f c) {
  return __builtin_amdgcn_wmma_f32_16x16x32_bf16(false, a, false, b, (short)0, c,
                                                 false, false);
}

// ---------------------------------------------------------------------------
// Elementwise f32 -> bf16 convert
// ---------------------------------------------------------------------------
__global__ __launch_bounds__(256)
void convert_f32_bf16(const float* __restrict__ src, unsigned short* __restrict__ dst,
                      size_t n) {
  size_t i = (size_t)blockIdx.x * blockDim.x + threadIdx.x;
  size_t stride = (size_t)gridDim.x * blockDim.x;
  for (; i < n; i += stride) dst[i] = f2bf(src[i]);
}

// ---------------------------------------------------------------------------
// LayerNorm over 512-wide rows (one wave per row, 8 rows per block).
// patch_mode==1: scatter result into the 2x2 im2col patch matrix [8192, 2048].
// ---------------------------------------------------------------------------
__global__ __launch_bounds__(256)
void ln512_kernel(const float* __restrict__ src, const float* __restrict__ g,
                  const float* __restrict__ bt, unsigned short* __restrict__ dst,
                  int nrows, int patch_mode) {
  const int wave = threadIdx.x >> 5;
  const int lane = threadIdx.x & 31;
  const int r = blockIdx.x * 8 + wave;
  if (r >= nrows) return;

  const float* x = src + (size_t)r * 512 + lane * 16;
  float vals[16];
#pragma unroll
  for (int i = 0; i < 4; ++i) {
    float4 t = reinterpret_cast<const float4*>(x)[i];
    vals[4 * i + 0] = t.x; vals[4 * i + 1] = t.y;
    vals[4 * i + 2] = t.z; vals[4 * i + 3] = t.w;
  }
  float s = 0.f, sq = 0.f;
#pragma unroll
  for (int i = 0; i < 16; ++i) { s += vals[i]; sq += vals[i] * vals[i]; }
#pragma unroll
  for (int m = 1; m < 32; m <<= 1) { s += __shfl_xor(s, m); sq += __shfl_xor(sq, m); }
  const float mean = s * (1.f / 512.f);
  const float var  = sq * (1.f / 512.f) - mean * mean;
  const float rstd = rsqrtf(var + 1e-5f);

  unsigned short* d;
  if (patch_mode) {
    const int b = r >> 12, n = r & 4095;
    const int yh = n >> 6, yw = n & 63;
    const int prow = b * 1024 + (yh >> 1) * 32 + (yw >> 1);
    const int slot = ((yh & 1) << 1) | (yw & 1);
    d = dst + (size_t)prow * 2048 + slot * 512 + lane * 16;
  } else {
    d = dst + (size_t)r * 512 + lane * 16;
  }
#pragma unroll
  for (int i = 0; i < 16; ++i) {
    const int c = lane * 16 + i;
    d[i] = f2bf((vals[i] - mean) * rstd * g[c] + bt[c]);
  }
}

// ---------------------------------------------------------------------------
// Generic bf16 WMMA GEMM: C[MxN] = A[MxK] * B[KxN]  (B stored K-major)
// Block = 4 waves; wave w computes rows [blk*128 + 32w, +32) x cols [n0, n0+64):
// two A fragments share every B fragment -> 8 WMMAs per 6 fragment loads.
// mode 0: f32 out (+ optional bias)        -> Cout
// mode 1: kv scatter: k -> kT[bh,d,Nk], v -> v[bh,Nk,d]   (bf16)
// mode 2: q scatter:  q -> q[bh,N,d]                       (bf16)
// ---------------------------------------------------------------------------
__global__ __launch_bounds__(128)
void gemm_wmma_kernel(const unsigned short* __restrict__ A,
                      const unsigned short* __restrict__ Bm,
                      int M, int N, int K,
                      const float* __restrict__ bias,
                      float* __restrict__ Cout,
                      unsigned short* __restrict__ outA,
                      unsigned short* __restrict__ outB,
                      int mode) {
  const int wave = threadIdx.x >> 5;
  const int lane = threadIdx.x & 31;
  const int m0 = blockIdx.y * 128 + wave * 32;
  const int n0 = blockIdx.x * 64;

  v8f acc[2][4] = {};
  const unsigned short* Ap = A + (size_t)m0 * K;
  for (int kk = 0; kk < K; kk += 32) {
    const v16bf a0 = load_frag_a16(Ap + kk, K);                   // rows m0..+15
    const v16bf a1 = load_frag_a16(Ap + (size_t)16 * K + kk, K);  // rows m0+16..+31
    const unsigned short* Bp = Bm + (size_t)kk * N + n0;
#pragma unroll
    for (int f = 0; f < 4; ++f) {
      const v16bf b = load_frag_b16(Bp + f * 16, N);
      acc[0][f] = wmma_bf16(a0, b, acc[0][f]);
      acc[1][f] = wmma_bf16(a1, b, acc[1][f]);
    }
  }

  const int rbase = (lane >> 4) << 3;   // C frag: lanes 0-15 rows 0-7, 16-31 rows 8-15
  const int nc    = lane & 15;

#pragma unroll
  for (int t = 0; t < 2; ++t) {
    const int mt = m0 + t * 16;
    if (mode == 0) {
#pragma unroll
      for (int f = 0; f < 4; ++f) {
        const int c = n0 + f * 16 + nc;
        const float bb = bias ? bias[c] : 0.f;
#pragma unroll
        for (int j = 0; j < 8; ++j)
          Cout[(size_t)(mt + rbase + j) * N + c] = acc[t][f][j] + bb;
      }
    } else if (mode == 1) {
#pragma unroll
      for (int f = 0; f < 4; ++f) {
        const int c = n0 + f * 16 + nc;
        const int kv2 = c >> 9, rem = c & 511, head = rem >> 6, dd = rem & 63;
#pragma unroll
        for (int j = 0; j < 8; ++j) {
          const int r = mt + rbase + j;
          const int b = r >> 10, m = r & 1023;
          const unsigned short val = f2bf(acc[t][f][j]);
          if (kv2 == 0)
            outA[((size_t)(b * 8 + head) * 64 + dd) * 1024 + m] = val;   // kT [bh,d,Nk]
          else
            outB[((size_t)(b * 8 + head) * 1024 + m) * 64 + dd] = val;   // v  [bh,Nk,d]
        }
      }
    } else {
#pragma unroll
      for (int f = 0; f < 4; ++f) {
        const int c = n0 + f * 16 + nc;
        const int head = c >> 6, dd = c & 63;
#pragma unroll
        for (int j = 0; j < 8; ++j) {
          const int r = mt + rbase + j;
          const int b = r >> 12, n = r & 4095;
          outA[((size_t)(b * 8 + head) * 4096 + n) * 64 + dd] = f2bf(acc[t][f][j]);
        }
      }
    }
  }
}

// ---------------------------------------------------------------------------
// Flash attention: q[bh,4096,64] x kT[bh,64,1024] -> softmax -> * v[bh,1024,64]
// 8 waves/block, each wave owns 16 q rows; key tiles of 64; P converted to the
// A-fragment layout via a per-wave LDS tile.
// ---------------------------------------------------------------------------
__global__ __launch_bounds__(256)
void attn_kernel(const unsigned short* __restrict__ qbuf,
                 const unsigned short* __restrict__ kT,
                 const unsigned short* __restrict__ vbuf,
                 unsigned short* __restrict__ attn_out) {
  __shared__ __align__(16) unsigned short p_lds[8 * 16 * 64];

  const int wave = threadIdx.x >> 5;
  const int lane = threadIdx.x & 31;
  const int bh = blockIdx.y;                 // 0..63
  const int b = bh >> 3, h = bh & 7;
  const int n0 = blockIdx.x * 128 + wave * 16;

  const unsigned short* qp = qbuf + ((size_t)bh * 4096 + n0) * 64;
  const unsigned short* kp = kT   + (size_t)bh * 64 * 1024;
  const unsigned short* vp = vbuf + (size_t)bh * 1024 * 64;
  unsigned short* pl = p_lds + wave * 16 * 64;

  const v16bf aq0 = load_frag_a16(qp + 0, 64);
  const v16bf aq1 = load_frag_a16(qp + 32, 64);

  v8f o[4] = {};
  float mrow[8], lrow[8];
#pragma unroll
  for (int j = 0; j < 8; ++j) { mrow[j] = -3.0e38f; lrow[j] = 0.f; }

  const float scale = 0.125f;                // 1/sqrt(64)
  const int rbase = (lane >> 4) << 3;
  const int nc = lane & 15;

  for (int m0 = 0; m0 < 1024; m0 += 64) {
    // ---- S = q @ k^T for 64 keys (4 C-fragments, K=64 split in 2) ----
    v8f s[4] = {};
#pragma unroll
    for (int f = 0; f < 4; ++f) {
      s[f] = wmma_bf16(aq0, load_frag_b16(kp + (m0 + f * 16), 1024), s[f]);
      s[f] = wmma_bf16(aq1, load_frag_b16(kp + 32 * 1024 + (m0 + f * 16), 1024), s[f]);
    }
    // ---- online softmax (row reductions across 16-lane halves) ----
#pragma unroll
    for (int j = 0; j < 8; ++j) {
      float mx = fmaxf(fmaxf(s[0][j], s[1][j]), fmaxf(s[2][j], s[3][j])) * scale;
      mx = fmaxf(mx, __shfl_xor(mx, 1));
      mx = fmaxf(mx, __shfl_xor(mx, 2));
      mx = fmaxf(mx, __shfl_xor(mx, 4));
      mx = fmaxf(mx, __shfl_xor(mx, 8));
      const float mo = mrow[j];
      const float mn = fmaxf(mo, mx);
      const float alpha = __expf(mo - mn);
      float psum = 0.f;
#pragma unroll
      for (int f = 0; f < 4; ++f) {
        const float pv = __expf(s[f][j] * scale - mn);
        s[f][j] = pv;
        psum += pv;
      }
      psum += __shfl_xor(psum, 1);
      psum += __shfl_xor(psum, 2);
      psum += __shfl_xor(psum, 4);
      psum += __shfl_xor(psum, 8);
      lrow[j] = lrow[j] * alpha + psum;
      mrow[j] = mn;
#pragma unroll
      for (int f = 0; f < 4; ++f) o[f][j] *= alpha;
    }
    // ---- P: C-layout -> A-layout via per-wave LDS tile (16 x 64) ----
#pragma unroll
    for (int f = 0; f < 4; ++f)
#pragma unroll
      for (int j = 0; j < 8; ++j)
        pl[(rbase + j) * 64 + f * 16 + nc] = f2bf(s[f][j]);

    const v16bf ap0 = load_frag_a16(pl + 0, 64);
    const v16bf ap1 = load_frag_a16(pl + 32, 64);
    // ---- O += P @ V ----
#pragma unroll
    for (int f2 = 0; f2 < 4; ++f2) {
      o[f2] = wmma_bf16(ap0, load_frag_b16(vp + (size_t)m0 * 64 + f2 * 16, 64), o[f2]);
      o[f2] = wmma_bf16(ap1, load_frag_b16(vp + (size_t)(m0 + 32) * 64 + f2 * 16, 64), o[f2]);
    }
  }

  // ---- epilogue: divide by l, store bf16 [B, N, h*d] ----
#pragma unroll
  for (int j = 0; j < 8; ++j) {
    const float inv = 1.f / lrow[j];
    const size_t row = (size_t)b * 4096 + n0 + rbase + j;
#pragma unroll
    for (int f2 = 0; f2 < 4; ++f2)
      attn_out[row * 512 + h * 64 + f2 * 16 + nc] = f2bf(o[f2][j] * inv);
  }
}

// ---------------------------------------------------------------------------
// Host orchestration
// ---------------------------------------------------------------------------
extern "C" void kernel_launch(void* const* d_in, const int* in_sizes, int n_in,
                              void* d_out, int out_size, void* d_ws, size_t ws_size,
                              hipStream_t stream) {
  const float* x       = (const float*)d_in[0];
  const float* y       = (const float*)d_in[1];
  const float* Wq      = (const float*)d_in[2];
  const float* Wkv     = (const float*)d_in[3];
  const float* Wproj   = (const float*)d_in[4];
  const float* bproj   = (const float*)d_in[5];
  const float* g_cross = (const float*)d_in[6];
  const float* b_cross = (const float*)d_in[7];
  const float* sr_w    = (const float*)d_in[8];
  const float* sr_b    = (const float*)d_in[9];
  const float* g_sr    = (const float*)d_in[10];
  const float* b_sr    = (const float*)d_in[11];
  (void)in_sizes; (void)n_in; (void)out_size; (void)ws_size;

  char* ws = (char*)d_ws;
  size_t off = 0;
  auto take = [&](size_t bytes) -> char* {
    char* p = ws + off;
    off = (off + bytes + 255) & ~(size_t)255;
    return p;
  };

  unsigned short* x_bf   = (unsigned short*)take(32768ull * 512 * 2);
  unsigned short* ypatch = (unsigned short*)take(8192ull * 2048 * 2);
  unsigned short* srw_bf = (unsigned short*)take(2048ull * 512 * 2);
  unsigned short* wq_bf  = (unsigned short*)take(512ull * 512 * 2);
  unsigned short* wkv_bf = (unsigned short*)take(512ull * 1024 * 2);
  unsigned short* wpj_bf = (unsigned short*)take(512ull * 512 * 2);
  float*          convo  = (float*)take(8192ull * 512 * 4);
  unsigned short* xsr    = (unsigned short*)take(8192ull * 512 * 2);
  unsigned short* qbuf   = (unsigned short*)take(64ull * 4096 * 64 * 2);
  unsigned short* kTb    = (unsigned short*)take(64ull * 64 * 1024 * 2);
  unsigned short* vbuf   = (unsigned short*)take(64ull * 1024 * 64 * 2);
  unsigned short* aout   = (unsigned short*)take(32768ull * 512 * 2);

  // bf16 conversions (weights kept K-major so B-fragments load contiguously)
  convert_f32_bf16<<<4096, 256, 0, stream>>>(x, x_bf, 32768ull * 512);
  convert_f32_bf16<<<256, 256, 0, stream>>>(Wq, wq_bf, 512ull * 512);
  convert_f32_bf16<<<256, 256, 0, stream>>>(Wkv, wkv_bf, 512ull * 1024);
  convert_f32_bf16<<<256, 256, 0, stream>>>(Wproj, wpj_bf, 512ull * 512);
  convert_f32_bf16<<<512, 256, 0, stream>>>(sr_w, srw_bf, 2048ull * 512);

  // LN(y) fused with 2x2 im2col -> [8192, 2048] patch matrix
  ln512_kernel<<<4096, 256, 0, stream>>>(y, g_cross, b_cross, ypatch, 32768, 1);

  // SR conv as GEMM: [8192,2048] x [2048,512] + sr_b -> f32
  gemm_wmma_kernel<<<dim3(8, 64), 128, 0, stream>>>(
      ypatch, srw_bf, 8192, 512, 2048, sr_b, convo, nullptr, nullptr, 0);

  // LN(g_sr, b_sr) -> bf16
  ln512_kernel<<<1024, 256, 0, stream>>>(convo, g_sr, b_sr, xsr, 8192, 0);

  // kv projection: [8192,512] x [512,1024] -> kT [bh,64,1024], v [bh,1024,64]
  gemm_wmma_kernel<<<dim3(16, 64), 128, 0, stream>>>(
      xsr, wkv_bf, 8192, 1024, 512, nullptr, nullptr, kTb, vbuf, 1);

  // q projection: [32768,512] x [512,512] -> q [bh,4096,64]
  gemm_wmma_kernel<<<dim3(8, 256), 128, 0, stream>>>(
      x_bf, wq_bf, 32768, 512, 512, nullptr, nullptr, qbuf, nullptr, 2);

  // flash attention
  attn_kernel<<<dim3(32, 64), 256, 0, stream>>>(qbuf, kTb, vbuf, aout);

  // output projection: [32768,512] x [512,512] + bproj -> f32 d_out
  gemm_wmma_kernel<<<dim3(8, 256), 128, 0, stream>>>(
      aout, wpj_bf, 32768, 512, 512, bproj, (float*)d_out, nullptr, nullptr, 0);
}
